// GatedRegression_22325240004852
// MI455X (gfx1250) — matrix operand order repaired and checked by reading, not compiled, not run on hardware
//
#include <hip/hip_runtime.h>
#include <hip/hip_bf16.h>
#include <math.h>

#define H 256
#define TWO_H 512
#define NUM_GRAPHS_C 2048
#define NUM_NODES_C 400000
#define ROWS_PER_WAVE 16
#define WAVES_PER_BLOCK 8
#define ROW_HALVES 520                      // 512 + 8 halves pad -> conflict-free b128 LDS reads
#define WAVE_SLAB (ROWS_PER_WAVE * ROW_HALVES)

typedef __attribute__((ext_vector_type(16))) _Float16 v16h;
typedef __attribute__((ext_vector_type(8)))  _Float16 v8h;
typedef __attribute__((ext_vector_type(8)))  float    v8f;
typedef __attribute__((ext_vector_type(4)))  float    v4f;

// fast activations: single TRANS32 ops (v_exp_f32 / v_rcp_f32) instead of
// libm's branchy expansions; keeps the VALU path short between WMMA bursts.
__device__ __forceinline__ float fast_sigmoid(float x) {
    return __builtin_amdgcn_rcpf(1.0f + __expf(-x));
}
__device__ __forceinline__ float fast_tanh(float x) {
    // tanh(x) = 1 - 2/(exp(2x)+1); exp->inf => t->1, exp->0 => t->-1
    return 1.0f - 2.0f * __builtin_amdgcn_rcpf(__expf(2.0f * x) + 1.0f);
}

// ---------------------------------------------------------------------------
// Repack gate_w [256,512] and trans_w [256,256] (f32, row-major) into f16 WMMA
// B-fragments: frag[fragIdx][lane][e], with B layout (16-bit, 32x16):
//   N = lane & 15,  K = (e&7) + 8*(lane>>4) + 16*(e>>3)
// gate frag index = j*16 + kb (j: col tile 0..15, kb: K tile 0..15)
// trans frag index = j*8 + kb  (kb: 0..7)
// ---------------------------------------------------------------------------
__global__ __launch_bounds__(128) void prep_weights_kernel(
    const float* __restrict__ gate_w, const float* __restrict__ trans_w,
    _Float16* __restrict__ gateB, _Float16* __restrict__ transB)
{
    const int wid  = (blockIdx.x * blockDim.x + threadIdx.x) >> 5;
    const int lane = threadIdx.x & 31;
    const int half = lane >> 4;
    const int lm   = lane & 15;
    if (wid < 256) {
        const int j = wid >> 4, kb = wid & 15;
        const float* src = gate_w + (size_t)(j * 16 + lm) * TWO_H + kb * 32 + 8 * half;
        v16h frag;
        #pragma unroll
        for (int e = 0; e < 16; ++e) frag[e] = (_Float16)src[e + 8 * (e >> 3)];
        *(v16h*)(gateB + ((size_t)wid * 32 + lane) * 16) = frag;
    } else if (wid < 384) {
        const int f = wid - 256;
        const int j = f >> 3, kb = f & 7;
        const float* src = trans_w + (size_t)(j * 16 + lm) * H + kb * 32 + 8 * half;
        v16h frag;
        #pragma unroll
        for (int e = 0; e < 16; ++e) frag[e] = (_Float16)src[e + 8 * (e >> 3)];
        *(v16h*)(transB + ((size_t)f * 32 + lane) * 16) = frag;
    }
}

__global__ __launch_bounds__(256) void zero_kernel(float* __restrict__ p, int n4)
{
    const int i = blockIdx.x * blockDim.x + threadIdx.x;
    if (i < n4) {
        v4f z = {0.f, 0.f, 0.f, 0.f};
        ((v4f*)p)[i] = z;
    }
}

// ---------------------------------------------------------------------------
// Fused: gate = sigmoid([emb|feat] @ gate_w^T + b), t = tanh(emb @ trans_w^T + b)
//        atomicAdd(graph_repr[gid[node]], gate * t)
// One wave per 16-node tile; f16 WMMA 16x16x32 with f32 accumulation.
// ---------------------------------------------------------------------------
__global__ __launch_bounds__(256) void fused_gate_kernel(
    const float* __restrict__ emb, const float* __restrict__ feat,
    const int* __restrict__ gids,
    const float* __restrict__ gate_b, const float* __restrict__ trans_b,
    const _Float16* __restrict__ gateB, const _Float16* __restrict__ transB,
    float* __restrict__ graph_out)
{
    extern __shared__ _Float16 lds[];
    const int wave = threadIdx.x >> 5;
    const int lane = threadIdx.x & 31;
    const int half = lane >> 4;
    const int lm   = lane & 15;
    const int tile = blockIdx.x * WAVES_PER_BLOCK + wave;   // 0..24999
    const int r0   = tile * ROWS_PER_WAVE;
    _Float16* slab = lds + wave * WAVE_SLAB;

    // ---- stage 16 rows of [emb(256) | feat(256)] into LDS as f16 ----
    #pragma unroll 4
    for (int row = 0; row < 16; ++row) {
        const float* se = emb  + (size_t)(r0 + row) * H + lane * 8;
        const float* sf = feat + (size_t)(r0 + row) * H + lane * 8;
        v4f e0 = *(const v4f*)se, e1 = *(const v4f*)(se + 4);
        v4f f0 = *(const v4f*)sf, f1 = *(const v4f*)(sf + 4);
        v8h he, hf;
        #pragma unroll
        for (int t = 0; t < 4; ++t) {
            he[t] = (_Float16)e0[t]; he[4 + t] = (_Float16)e1[t];
            hf[t] = (_Float16)f0[t]; hf[4 + t] = (_Float16)f1[t];
        }
        *(v8h*)(slab + row * ROW_HALVES + lane * 8)     = he;
        *(v8h*)(slab + row * ROW_HALVES + H + lane * 8) = hf;
    }
    // same-wave LDS RAW: make ds stores visible before fragment loads
    asm volatile("s_wait_dscnt 0" ::: "memory");

    // graph ids for the rows this lane's accumulators map to (M = v + 8*half)
    int gidv[8];
    #pragma unroll
    for (int v = 0; v < 8; ++v) gidv[v] = gids[r0 + v + 8 * half];

    // A layout (16-bit, 16x32): M = lane&15, K = (e&7) + 8*(lane>>4) + 16*(e>>3)
    const _Float16* arow = slab + lm * ROW_HALVES;

    for (int j = 0; j < 16; ++j) {             // output column tile
        v8f gacc = {};
        #pragma unroll
        for (int kb = 0; kb < 16; ++kb) {      // gate: K = 512
            const _Float16* ka = arow + kb * 32 + 8 * half;
            v8h lo = *(const v8h*)(ka);
            v8h hi = *(const v8h*)(ka + 16);
            v16h a = __builtin_shufflevector(lo, hi, 0,1,2,3,4,5,6,7,8,9,10,11,12,13,14,15);
            v16h b = *(const v16h*)(gateB + ((size_t)(j * 16 + kb) * 32 + lane) * 16);
            gacc = __builtin_amdgcn_wmma_f32_16x16x32_f16(
                false, a, false, b, (short)0, gacc, false, false);
        }
        v8f tacc = {};
        #pragma unroll
        for (int kb = 0; kb < 8; ++kb) {       // trans: K = 256 (emb half of slab)
            const _Float16* ka = arow + kb * 32 + 8 * half;
            v8h lo = *(const v8h*)(ka);
            v8h hi = *(const v8h*)(ka + 16);
            v16h a = __builtin_shufflevector(lo, hi, 0,1,2,3,4,5,6,7,8,9,10,11,12,13,14,15);
            v16h b = *(const v16h*)(transB + ((size_t)(j * 8 + kb) * 32 + lane) * 16);
            tacc = __builtin_amdgcn_wmma_f32_16x16x32_f16(
                false, a, false, b, (short)0, tacc, false, false);
        }
        // C/D layout: element v -> row M = v + 8*half, col N = lane&15
        const int col = j * 16 + lm;
        const float gbias = gate_b[col];
        const float tbias = trans_b[col];
        #pragma unroll
        for (int v = 0; v < 8; ++v) {
            float g = fast_sigmoid(gacc[v] + gbias);
            float t = fast_tanh(tacc[v] + tbias);
            atomicAdd(&graph_out[(size_t)gidv[v] * H + col], g * t);
        }
    }
}

// ---------------------------------------------------------------------------
// prediction[g] = graph_repr[g] . out_w + out_b    (one wave per graph)
// ---------------------------------------------------------------------------
__global__ __launch_bounds__(256) void out_proj_kernel(
    const float* __restrict__ graph_repr, const float* __restrict__ out_w,
    const float* __restrict__ out_b, float* __restrict__ pred)
{
    const int g    = (blockIdx.x * blockDim.x + threadIdx.x) >> 5;
    const int lane = threadIdx.x & 31;
    if (g >= NUM_GRAPHS_C) return;
    const float* row = graph_repr + (size_t)g * H;
    float s = 0.f;
    #pragma unroll
    for (int i = 0; i < H / 32; ++i)
        s += row[lane + i * 32] * out_w[lane + i * 32];
    #pragma unroll
    for (int o = 16; o > 0; o >>= 1)
        s += __shfl_down(s, o, 32);
    if (lane == 0) pred[g] = s + out_b[0];
}

extern "C" void kernel_launch(void* const* d_in, const int* in_sizes, int n_in,
                              void* d_out, int out_size, void* d_ws, size_t ws_size,
                              hipStream_t stream)
{
    const float* emb     = (const float*)d_in[0];   // [400000, 256]
    const float* feat    = (const float*)d_in[1];   // [400000, 256]
    const int*   gids    = (const int*)d_in[2];     // [400000]
    const float* gate_w  = (const float*)d_in[4];   // [256, 512]
    const float* gate_b  = (const float*)d_in[5];   // [256]
    const float* trans_w = (const float*)d_in[6];   // [256, 256]
    const float* trans_b = (const float*)d_in[7];   // [256]
    const float* out_w   = (const float*)d_in[8];   // [256]
    const float* out_b   = (const float*)d_in[9];   // [1]

    float* pred       = (float*)d_out;              // [2048]
    float* graph_repr = pred + NUM_GRAPHS_C;        // [2048, 256]

    _Float16* gateB  = (_Float16*)d_ws;               // 256 frags * 32 * 16 halves (256 KB)
    _Float16* transB = gateB + (size_t)256 * 32 * 16; // 128 frags (128 KB)

    prep_weights_kernel<<<96, 128, 0, stream>>>(gate_w, trans_w, gateB, transB);
    zero_kernel<<<(NUM_GRAPHS_C * H / 4 + 255) / 256, 256, 0, stream>>>(
        graph_repr, NUM_GRAPHS_C * H / 4);

    const size_t shmem = (size_t)WAVES_PER_BLOCK * WAVE_SLAB * sizeof(_Float16); // 130 KB
    fused_gate_kernel<<<NUM_NODES_C / (ROWS_PER_WAVE * WAVES_PER_BLOCK), 256, shmem, stream>>>(
        emb, feat, gids, gate_b, trans_b, gateB, transB, graph_repr);

    out_proj_kernel<<<(NUM_GRAPHS_C * 32) / 256, 256, 0, stream>>>(
        graph_repr, out_w, out_b, pred);
}
